// LGCR_66374424592494
// MI455X (gfx1250) — matrix operand order repaired and checked by reading, not compiled, not run on hardware
//
#include <hip/hip_runtime.h>
#include <hip/hip_bf16.h>

// Problem dims (fixed by the reference)
#define B_ 16
#define C_ 512
#define H_ 64
#define W_ 64
#define R_ 64

typedef __attribute__((ext_vector_type(16))) __bf16       v16bf;
typedef __attribute__((ext_vector_type(8)))  float        v8f;
typedef __attribute__((ext_vector_type(4)))  unsigned int u32x4;

union V16U { u32x4 q[2]; v16bf v; };

__device__ __forceinline__ unsigned short f2bf(float f) {
  unsigned u = __builtin_bit_cast(unsigned, f);
  u += 0x7FFFu + ((u >> 16) & 1u);           // round-to-nearest-even
  return (unsigned short)(u >> 16);
}

// ---------------------------------------------------------------------------
// Kernel 1: per-(b,c) plane sums. Deterministic (no float atomics):
//   gapc[b,c]          = sum_{h,w} x
//   partW[(b,c), w]    = sum_h x   (folded over c later)
//   partH[(b,c), h]    = sum_w x
// ---------------------------------------------------------------------------
__global__ __launch_bounds__(256) void k_gap(const float* __restrict__ x,
                                             float* __restrict__ gapc,
                                             float* __restrict__ partW,
                                             float* __restrict__ partH) {
  const int c = blockIdx.x, b = blockIdx.y, t = threadIdx.x;
  const float* p = x + ((size_t)(b * C_ + c)) * (H_ * W_);

  __shared__ float lw[64][4];
  __shared__ float lh[64][4];
  __shared__ float sRowW[64];

  // pass 1: column(w) partials, coalesced rows
  {
    int w = t & 63, hb = t >> 6;
    float tw = 0.f;
#pragma unroll
    for (int k = 0; k < 16; ++k) tw += p[(hb + 4 * k) * W_ + w];
    lw[w][hb] = tw;
  }
  // pass 2: row(h) partials (plane is L0-resident after pass 1)
  {
    int hh = t & 63, wb = t >> 6;
    float th = 0.f;
#pragma unroll
    for (int k = 0; k < 16; ++k) th += p[hh * W_ + wb + 4 * k];
    lh[hh][wb] = th;
  }
  __syncthreads();
  if (t < 64) {
    float pw = lw[t][0] + lw[t][1] + lw[t][2] + lw[t][3];
    float ph = lh[t][0] + lh[t][1] + lh[t][2] + lh[t][3];
    partW[((size_t)(b * C_ + c)) * 64 + t] = pw;
    partH[((size_t)(b * C_ + c)) * 64 + t] = ph;
    sRowW[t] = pw;
  }
  __syncthreads();
  if (t == 0) {
    float s = 0.f;
#pragma unroll
    for (int i = 0; i < 64; ++i) s += sRowW[i];
    gapc[b * C_ + c] = s;
  }
}

// ---------------------------------------------------------------------------
// Kernel 2: fold partials over c (fixed order -> deterministic)
// ---------------------------------------------------------------------------
__global__ __launch_bounds__(256) void k_reduce(const float* __restrict__ partW,
                                                const float* __restrict__ partH,
                                                float* __restrict__ gapw,
                                                float* __restrict__ gaph) {
  int i = blockIdx.x * 256 + threadIdx.x;     // 2048 total
  if (i >= 2 * B_ * 64) return;
  int sel = i >> 10;                          // 0 -> gapw, 1 -> gaph
  int j = i & 1023;
  int b = j >> 6, w = j & 63;
  const float* src = sel ? partH : partW;
  float s = 0.f;
  for (int c = 0; c < C_; ++c) s += src[((size_t)(b * C_ + c)) * 64 + w];
  (sel ? gaph : gapw)[j] = s;
}

// ---------------------------------------------------------------------------
// Kernel 3: the three conv1d(k=3,pad=1)+PReLU branches.
//   cA[b][c][r]   (bf16, r-contiguous: matches WMMA A-operand chunking)
//   hO[b][r][h], wO[b][r][w]  (f32)
// ---------------------------------------------------------------------------
__global__ __launch_bounds__(256) void k_branch(
    const float* __restrict__ gapc, const float* __restrict__ gapw,
    const float* __restrict__ gaph,
    const float* __restrict__ w_c, const float* __restrict__ a_c,
    const float* __restrict__ w_w, const float* __restrict__ a_w,
    const float* __restrict__ w_h, const float* __restrict__ a_h,
    unsigned short* __restrict__ cA, float* __restrict__ hO,
    float* __restrict__ wO) {
  int idx = blockIdx.x * 256 + threadIdx.x;
  if (idx < B_ * C_ * R_) {                   // channel branch
    int b = idx >> 15, rem = idx & 32767;
    int c = rem >> 6, r = rem & 63;
    const float* g = gapc + b * C_;
    const float sc = 1.f / (H_ * W_);
    float gm1 = (c > 0) ? g[c - 1] : 0.f;
    float g0 = g[c];
    float gp1 = (c < C_ - 1) ? g[c + 1] : 0.f;
    float s = (w_c[r * 3] * gm1 + w_c[r * 3 + 1] * g0 + w_c[r * 3 + 2] * gp1) * sc;
    s = (s >= 0.f) ? s : a_c[0] * s;
    cA[idx] = f2bf(s);                        // idx == ((b*C_+c)*R_+r)
  } else {
    int j = idx - B_ * C_ * R_;
    if (j >= 2 * B_ * R_ * 64) return;
    int branch = j >> 16;                     // 0 -> width, 1 -> height
    j &= 65535;
    int b = j >> 12, r = (j >> 6) & 63, p = j & 63;
    const float* g = (branch ? gaph : gapw) + b * 64;
    const float* wgt = branch ? w_h : w_w;
    float a = branch ? a_h[0] : a_w[0];
    const float sc = 1.f / (C_ * 64);
    float gm1 = (p > 0) ? g[p - 1] : 0.f;
    float g0 = g[p];
    float gp1 = (p < 63) ? g[p + 1] : 0.f;
    float s = (wgt[r * 3] * gm1 + wgt[r * 3 + 1] * g0 + wgt[r * 3 + 2] * gp1) * sc;
    s = (s >= 0.f) ? s : a * s;
    (branch ? hO : wO)[(b * R_ + r) * 64 + p] = s;
  }
}

// ---------------------------------------------------------------------------
// Kernel 4: fused CP-einsum (bf16 WMMA GEMM) + channel softmax + conv3d + mix.
// One workgroup per (b, h, 16-wide w tile); 4 waves, 8 C-tiles each.
// ---------------------------------------------------------------------------
__global__ __launch_bounds__(128) void k_main(
    const float* __restrict__ x, const unsigned short* __restrict__ cA,
    const float* __restrict__ hO, const float* __restrict__ wO,
    const float* __restrict__ w3d, float* __restrict__ out) {
  const int wt = blockIdx.x, h = blockIdx.y, b = blockIdx.z;
  const int w0 = wt * 16;
  const int t = threadIdx.x;

  __shared__ __align__(16) unsigned short sB[16 * 72];  // hw^T [w][r], padded
  __shared__ float sCp[C_ * 16];                        // logits -> exp()
  __shared__ float sPart[8][16];
  __shared__ float sMax[16];
  __shared__ float sInv[16];
  __shared__ float sW3[27];

  if (t < 27) sW3[t] = w3d[t];
  // B^T tile: sB[w][r] = bf16( h_out[b,r,h] * w_out[b,r,w0+w] )
#pragma unroll
  for (int k = 0; k < 8; ++k) {
    int e = t + 128 * k;
    int w = e >> 6, r = e & 63;
    float hv = hO[(b * R_ + r) * 64 + h];
    float wv = wO[(b * R_ + r) * 64 + w0 + w];
    sB[w * 72 + r] = f2bf(hv * wv);
  }
  __syncthreads();

  const int lane = t & 31;
  const int ln = lane & 15;
  const int hi = lane >> 4;                   // lane 16..31 half
  const int wv_id = t >> 5;                   // wave id 0..3

  // B operand (this kernel's k order, same as A): lane n holds N=n; lo lanes K=0..15,
  // hi lanes K=16..31 (V0..7, two bf16 per VGPR). Contiguous in sB rows.
  const unsigned short* bRow = sB + ln * 72 + hi * 16;
  V16U B0, B1;
  B0.q[0] = *(const u32x4*)(bRow);            // K block 0..31
  B0.q[1] = *(const u32x4*)(bRow + 8);
  B1.q[0] = *(const u32x4*)(bRow + 32);       // K block 32..63
  B1.q[1] = *(const u32x4*)(bRow + 40);

#pragma unroll
  for (int ct = 0; ct < 8; ++ct) {
    int c0 = (wv_id * 8 + ct) * 16;
    // A operand: lane row M=c0+ln; lo lanes K 0..15, hi lanes K 16..31.
    const unsigned short* aBase =
        cA + ((size_t)(b * C_ + c0 + ln)) * R_ + hi * 16;
    V16U A0, A1;
    A0.q[0] = *(const u32x4*)(aBase);         // K 0..7  (or 16..23)
    A0.q[1] = *(const u32x4*)(aBase + 8);     // K 8..15 (or 24..31)
    A1.q[0] = *(const u32x4*)(aBase + 32);    // K 32..39 (or 48..55)
    A1.q[1] = *(const u32x4*)(aBase + 40);    // K 40..47 (or 56..63)

    v8f acc = {};
    acc = __builtin_amdgcn_wmma_f32_16x16x32_bf16(false, A0.v, false, B0.v,
                                                  (short)0, acc, false, false);
    acc = __builtin_amdgcn_wmma_f32_16x16x32_bf16(false, A1.v, false, B1.v,
                                                  (short)0, acc, false, false);
    // C/D layout: VGPR i -> row i (lo lanes) / i+8 (hi lanes), col = lane%16
#pragma unroll
    for (int i = 0; i < 8; ++i)
      sCp[(c0 + i + hi * 8) * 16 + ln] = acc[i];
  }
  __syncthreads();

  // softmax over C per column (deterministic tree: 8 groups then 16 lanes)
  const int col = t & 15, grp = t >> 4;
  float m = -3.4e38f;
  for (int c = grp; c < C_; c += 8) m = fmaxf(m, sCp[c * 16 + col]);
  sPart[grp][col] = m;
  __syncthreads();
  if (t < 16) {
    float mm = sPart[0][t];
#pragma unroll
    for (int g = 1; g < 8; ++g) mm = fmaxf(mm, sPart[g][t]);
    sMax[t] = mm;
  }
  __syncthreads();
  float s = 0.f;
  const float cmax = sMax[col];
  for (int c = grp; c < C_; c += 8) {
    float e = __expf(sCp[c * 16 + col] - cmax);
    sCp[c * 16 + col] = e;
    s += e;
  }
  sPart[grp][col] = s;
  __syncthreads();
  if (t < 16) {
    float ss = 0.f;
#pragma unroll
    for (int g = 0; g < 8; ++g) ss += sPart[g][t];
    sInv[t] = 1.f / ss;
  }
  __syncthreads();

  // fused 3x3x3 conv over (C,H,W) + res*softmax + x  (x is L2-resident)
  const float* xb = x + (size_t)b * C_ * H_ * W_;
  float* ob = out + (size_t)b * C_ * H_ * W_;
#pragma unroll 2
  for (int k = 0; k < 64; ++k) {
    int idx = t + k * 128;
    int c = idx >> 4, w = idx & 15;
    int ww = w0 + w;
    if (c + 2 < C_)  // pull next-but-one channel plane toward L0/L2
      __builtin_prefetch(xb + (size_t)(c + 2) * (H_ * W_) + h * W_ + ww, 0, 1);
    float res = 0.f;
#pragma unroll
    for (int dc = -1; dc <= 1; ++dc) {
      int cc = c + dc;
      if (cc < 0 || cc >= C_) continue;
      const float* xp = xb + (size_t)cc * (H_ * W_);
#pragma unroll
      for (int dh = -1; dh <= 1; ++dh) {
        int hh = h + dh;
        if (hh < 0 || hh >= H_) continue;
        const float* xr = xp + hh * W_;
#pragma unroll
        for (int dw = -1; dw <= 1; ++dw) {
          int w2 = ww + dw;
          if (w2 < 0 || w2 >= W_) continue;
          res += sW3[(dc + 1) * 9 + (dh + 1) * 3 + (dw + 1)] * xr[w2];
        }
      }
    }
    float soft = sCp[c * 16 + w] * sInv[w];
    size_t o = (size_t)c * (H_ * W_) + h * W_ + ww;
    ob[o] = res * soft + xb[o];
  }
}

// ---------------------------------------------------------------------------
extern "C" void kernel_launch(void* const* d_in, const int* in_sizes, int n_in,
                              void* d_out, int out_size, void* d_ws,
                              size_t ws_size, hipStream_t stream) {
  const float* x   = (const float*)d_in[0];
  const float* w_c = (const float*)d_in[1];
  const float* a_c = (const float*)d_in[2];
  const float* w_w = (const float*)d_in[3];
  const float* a_w = (const float*)d_in[4];
  const float* w_h = (const float*)d_in[5];
  const float* a_h = (const float*)d_in[6];
  const float* w3d = (const float*)d_in[7];
  float* out = (float*)d_out;

  float* ws = (float*)d_ws;
  float* gapc  = ws;                 // 16*512                  =   8192 f
  float* gapw  = ws + 8192;          // 16*64                   =   1024 f
  float* gaph  = ws + 9216;          // 16*64                   =   1024 f
  float* hO    = ws + 10240;         // 16*64*64                =  65536 f
  float* wO    = ws + 75776;         // 16*64*64                =  65536 f
  float* partW = ws + 141312;        // 16*512*64               = 524288 f
  float* partH = ws + 665600;        // 16*512*64               = 524288 f
  unsigned short* cA =
      (unsigned short*)(ws + 1189888);  // 16*512*64 bf16 (1 MB), 16B-aligned

  k_gap<<<dim3(C_, B_), 256, 0, stream>>>(x, gapc, partW, partH);
  k_reduce<<<8, 256, 0, stream>>>(partW, partH, gapw, gaph);
  k_branch<<<(B_ * C_ * R_ + 2 * B_ * R_ * 64 + 255) / 256, 256, 0, stream>>>(
      gapc, gapw, gaph, w_c, a_c, w_w, a_w, w_h, a_h, cA, hO, wO);
  k_main<<<dim3(W_ / 16, H_, B_), 128, 0, stream>>>(x, cA, hO, wO, w3d, out);
}
